// DSModelMultiQ_59648505806947
// MI455X (gfx1250) — matrix operand order repaired and checked by reading, not compiled, not run on hardware
//
#include <hip/hip_runtime.h>
#include <math.h>

typedef __attribute__((ext_vector_type(16))) _Float16 v16h;
typedef __attribute__((ext_vector_type(8)))  float    v8f;

#define F   64
#define R   256
#define KC  10          // classes
#define L   1024        // literals
#define LPR 4
#define EPSV 1e-12f

// ---------------------------------------------------------------------------
// Kernel 1: softmax over rule_mass_params rows, build logAB (256 x 16, f16),
// and emit it directly as V_WMMA_F32_16X16X32_F16 B fragments in this code's contiguous k order (k = 16h + s):
//   bFrag[chunk][lane][half] , chunk=0..7 covers K (=rule) 32 at a time.
//   lanes 0..15 : N = lane,    halves h -> K = chunk*32 + h
//   lanes 16..31: N = lane-16, halves h -> K = chunk*32 + 16 + h
// ---------------------------------------------------------------------------
__global__ void ds_prep_kernel(const float* __restrict__ rmp,
                               _Float16* __restrict__ bFrag) {
    __shared__ float s_logAB[R][16];
    const int tid = threadIdx.x;      // 256 threads, one rule each
    {
        const int r = tid;
        float p[KC + 1];
        float mx = -1e30f;
        #pragma unroll
        for (int k = 0; k <= KC; ++k) { p[k] = rmp[r * (KC + 1) + k]; mx = fmaxf(mx, p[k]); }
        float den = 0.f;
        #pragma unroll
        for (int k = 0; k <= KC; ++k) { p[k] = __expf(p[k] - mx); den += p[k]; }
        const float inv = 1.f / den;
        const float mO  = p[KC] * inv;
        #pragma unroll
        for (int k = 0; k < KC; ++k)
            s_logAB[r][k] = __logf(p[k] * inv + mO + EPSV);
        s_logAB[r][KC] = __logf(mO + EPSV);
        #pragma unroll
        for (int k = KC + 1; k < 16; ++k) s_logAB[r][k] = 0.f;  // padded cols
    }
    __syncthreads();
    // scatter into fragment layout: flat idx = (chunk*32 + lane)*16 + half
    for (int idx = tid; idx < 8 * 32 * 16; idx += blockDim.x) {
        const int c = idx >> 9;
        const int l = (idx >> 4) & 31;
        const int h = idx & 15;
        const int k = c * 32 + ((l < 16) ? h : (16 + h));   // rule index
        const int n = l & 15;                               // output column
        bFrag[idx] = (_Float16)s_logAB[k][n];
    }
}

// ---------------------------------------------------------------------------
// Kernel 2: 128 threads = 4 wave32; each wave produces a 16-sample x 16-col
// output tile via 8x v_wmma_f32_16x16x32_f16 over the 256-rule K dimension.
// ---------------------------------------------------------------------------
__global__ void __launch_bounds__(128)
ds_main_kernel(const float* __restrict__ X,
               const int*   __restrict__ lit_feat,
               const int*   __restrict__ lit_op,
               const float* __restrict__ lit_val,
               const int*   __restrict__ rule_len,
               const _Float16* __restrict__ bFrag,
               float* __restrict__ out,
               int nSamples) {
    __shared__ int      s_code[L];           // feat | (op<<8)
    __shared__ float    s_val[L];
    __shared__ int      s_rlen[R];
    __shared__ float    s_X[4][16][F];       // 16 KB
    __shared__ _Float16 s_act[4][16][R];     // 32 KB

    const int tid  = threadIdx.x;
    const int wave = tid >> 5;
    const int lane = tid & 31;

    // literal + rule tables -> LDS
    for (int i = tid; i < L; i += blockDim.x) {
        s_code[i] = (lit_feat[i] & 0xff) | (lit_op[i] << 8);
        s_val[i]  = lit_val[i];
    }
    for (int i = tid; i < R; i += blockDim.x) s_rlen[i] = rule_len[i];

    // stage this wave's 16 X rows (b128 loads)
    const int base = blockIdx.x * 64 + wave * 16;
    for (int t = lane; t < 16 * (F / 4); t += 32) {
        const int row = t >> 4;
        const int c4  = t & 15;
        int n = base + row; n = (n < nSamples) ? n : (nSamples - 1);
        const float4 v = *(const float4*)(X + (size_t)n * F + c4 * 4);
        *(float4*)(&s_X[wave][row][c4 * 4]) = v;
    }
    __syncthreads();

    // evaluate 1024 literals -> 256 rule activations for each of 16 samples.
    // lane (s = lane&15, h = lane>>4) handles rules [h*128, h*128+128).
    {
        const int s = lane & 15;
        const int h = lane >> 4;
        for (int j = 0; j < 128; ++j) {
            const int r = h * 128 + j;
            int cnt = 0;
            #pragma unroll
            for (int t = 0; t < LPR; ++t) {
                const int   li   = r * LPR + t;
                const int   code = s_code[li];
                const float x    = s_X[wave][s][code & 0xff];
                const float v    = s_val[li];
                const int   op   = code >> 8;
                const bool  tr   = (op == 0) ? (x == v) : ((op == 1) ? (x < v) : (x > v));
                cnt += tr ? 1 : 0;
            }
            s_act[wave][s][r] = (_Float16)((cnt == s_rlen[r]) ? 1.0f : 0.0f);
        }
    }
    __syncthreads();

    // ---- WMMA: D(16x16) = active(16x256) @ logAB(256x16), f32 accumulate ----
    v8f acc = {};
    const int m   = lane & 15;           // A row
    const int hi8 = (lane >> 4) * 16;    // K sub-offset per A layout (contiguous: k = 16h + s)
    #pragma unroll
    for (int c = 0; c < 8; ++c) {
        v16h a;
        const _Float16* ap0 = &s_act[wave][m][c * 32 + hi8];        // halves 0..7
        const _Float16* ap1 = &s_act[wave][m][c * 32 + 8 + hi8];    // halves 8..15
        #pragma unroll
        for (int e = 0; e < 8; ++e) { a[e] = ap0[e]; a[8 + e] = ap1[e]; }
        const v16h b = *(const v16h*)(bFrag + ((c * 32 + lane) << 4));
        acc = __builtin_amdgcn_wmma_f32_16x16x32_f16(
            /*neg_a=*/false, a, /*neg_b=*/false, b,
            /*c_mod=*/(short)0, acc, /*reuse_a=*/false, /*reuse_b=*/false);
    }

    // ---- epilogue: per row M = g + 8*(lane>=16), col = lane&15 ----
    const int col  = lane & 15;
    const int half = lane >> 4;
    #pragma unroll
    for (int g = 0; g < 8; ++g) {
        const float e = __expf(acc[g]);
        const float q = __shfl(e, 10, 16);                 // column 10 = overflow mass
        float cls = (col < KC) ? (e - q) : 0.f;
        float sum = cls;
        #pragma unroll
        for (int off = 1; off < 16; off <<= 1) sum += __shfl_xor(sum, off, 16);
        const float total = fmaxf(sum + q, EPSV);
        const float ov = (col < KC) ? (e - q) / total : q / total;
        const int M = g + half * 8;
        const int n = base + M;
        if (n < nSamples && col <= KC) out[(size_t)n * (KC + 1) + col] = ov;
    }
}

// ---------------------------------------------------------------------------
extern "C" void kernel_launch(void* const* d_in, const int* in_sizes, int n_in,
                              void* d_out, int out_size, void* d_ws, size_t ws_size,
                              hipStream_t stream) {
    const float* X     = (const float*)d_in[0];
    const float* rmp   = (const float*)d_in[1];
    const int*   feat  = (const int*)d_in[2];
    const int*   opc   = (const int*)d_in[3];
    const float* lval  = (const float*)d_in[4];
    // d_in[5] = lit2rule (arange/LPR by construction), d_in[6] = rule_len
    const int*   rlen  = (const int*)d_in[6];
    float*       out   = (float*)d_out;
    _Float16*    bFrag = (_Float16*)d_ws;   // 8*32*16 halves = 8 KB

    const int nSamples = in_sizes[0] / F;

    ds_prep_kernel<<<1, 256, 0, stream>>>(rmp, bFrag);
    const int blocks = (nSamples + 63) / 64;
    ds_main_kernel<<<blocks, 128, 0, stream>>>(X, feat, opc, lval, rlen, bFrag,
                                               out, nSamples);
}
